// TMPNN_52664888984167
// MI455X (gfx1250) — compile-verified, matrix operand order checked by reading, not run on hardware
//
#include <hip/hip_runtime.h>

// CDNA5 / gfx1250, wave32. WMMA bf16 16x16x32, f32 accumulate.
typedef __attribute__((ext_vector_type(16))) __bf16       v16bf;
typedef __attribute__((ext_vector_type(8)))  float        v8f;
typedef __attribute__((ext_vector_type(8)))  unsigned int v8u;

#define N_FEAT   32
#define STEPS    7
#define N_BTILES 66            // 2 linear (W rows 1..32) + 64 quad (32 K-chunks x 2 N-halves)
#define PACK_ELEMS (N_BTILES * 512)  // 512 bf16 per 32x16 B tile -> 67584 B in d_ws

// f32 -> bf16 round-to-nearest-even (bits)
__device__ __forceinline__ unsigned f2bf_rne(float f) {
  unsigned u = __builtin_bit_cast(unsigned, f);
  return (u + 0x7FFFu + ((u >> 16) & 1u)) >> 16;
}

// K index held by fragment slot s (0..15) for lane-half h (lane>>4), per the
// CDNA5 16-bit A-matrix 16x32 layout (ISA 7.12.2); B 32x16 assumed mirrored
// (lane&15 -> N, same K split per half).
__device__ __forceinline__ int kmap(int s, int h) {
  return (s & 7) + 8 * h + 16 * (s >> 3);
}

// ---------------------------------------------------------------------------
// Pack W (1057x32 f32, row-major) into B-fragment-ordered bf16 in d_ws.
// Tile T: T<2 -> linear term (W rows 1..32), N-half = T.
//         T>=2 -> quad chunk c=(T-2)>>1 (W rows 33+32c .. +31), N-half=(T-2)&1.
// Element order inside a tile: lane L (0..31) * 16 + slot s, so a lane's whole
// v16bf B operand is one contiguous 32-byte (2x b128) load.
// ---------------------------------------------------------------------------
__global__ void tmpnn_pack_w(const float* __restrict__ W,
                             unsigned short* __restrict__ wb) {
  int idx = blockIdx.x * blockDim.x + threadIdx.x;
  if (idx >= PACK_ELEMS) return;
  int T   = idx >> 9;
  int rem = idx & 511;
  int L   = rem >> 4;
  int s   = rem & 15;
  int h   = L >> 4;
  int k   = kmap(s, h);
  int row, nh;
  if (T < 2) { row = 1 + k; nh = T; }
  else       { int c = (T - 2) >> 1; nh = (T - 2) & 1; row = 33 + 32 * c + k; }
  int col = nh * 16 + (L & 15);
  wb[idx] = (unsigned short)f2bf_rne(W[row * N_FEAT + col]);
}

// ---------------------------------------------------------------------------
// Main kernel: each wave owns one 16-row tile for all 7 steps.
// Every lane keeps the full f32 row (row = lane&15; halves duplicated, which
// is exactly what the A-fragment layout wants). Per step:
//   acc(N-col) = bias  ;  acc += A_lin @ W1  ;  for c: acc += (x_c*A_lin) @ W2_c
// then transpose D through per-wave LDS scratch and do the residual add.
// ---------------------------------------------------------------------------
__global__ __launch_bounds__(256)
void tmpnn_main(const float* __restrict__ x, const float* __restrict__ W,
                const unsigned short* __restrict__ wb, float* __restrict__ out,
                int n) {
  __shared__ __align__(16) float tbuf[8 * 16 * N_FEAT];  // 16 KB, per-wave slices

  const int lane = threadIdx.x & 31;
  const int wave = threadIdx.x >> 5;
  const int hlf  = lane >> 4;
  const int lrow = lane & 15;

  long ntiles = (long)(n >> 4);
  long tile = (long)blockIdx.x * 8 + wave;
  if (tile >= ntiles) tile = ntiles - 1;  // clamp (no divergence/EXEC holes, no barrier deadlock)
  const long rowBase = tile * 16;

  // Bias row folded into accumulator init: element (M, N) of acc depends only on N.
  const float b0s = W[lrow];        // N = lrow
  const float b1s = W[16 + lrow];   // N = 16 + lrow

  // Full row in registers (8 x b128 loads).
  float xr[N_FEAT];
  const float4* xp = (const float4*)(x + (rowBase + lrow) * N_FEAT);
#pragma unroll
  for (int i = 0; i < 8; ++i) {
    float4 v = xp[i];
    xr[4 * i + 0] = v.x; xr[4 * i + 1] = v.y;
    xr[4 * i + 2] = v.z; xr[4 * i + 3] = v.w;
  }

  float* tw = tbuf + wave * (16 * N_FEAT);
  const v16bf* wbv = (const v16bf*)wb;  // tile t, lane l -> wbv[t*32 + l]

  for (int step = 0; step < STEPS; ++step) {
    // Packed bf16 linear A-fragment from this lane's K subset.
    v8u au;
#pragma unroll
    for (int r = 0; r < 8; ++r) {
      int k0 = 2 * (r & 3) + 8 * hlf + 16 * (r >> 2);   // = kmap(2r, hlf)
      au[r] = f2bf_rne(xr[k0]) | (f2bf_rne(xr[k0 + 1]) << 16);
    }
    v16bf alin = __builtin_bit_cast(v16bf, au);

    v8f acc0, acc1;
#pragma unroll
    for (int r = 0; r < 8; ++r) { acc0[r] = b0s; acc1[r] = b1s; }

    // Linear term: K = 32 in one WMMA per N-half.
    {
      v16bf bl0 = wbv[0 * 32 + lane];
      v16bf bl1 = wbv[1 * 32 + lane];
      acc0 = __builtin_amdgcn_wmma_f32_16x16x32_bf16(false, alin, false, bl0,
                                                     (short)0, acc0, false, false);
      acc1 = __builtin_amdgcn_wmma_f32_16x16x32_bf16(false, alin, false, bl1,
                                                     (short)0, acc1, false, false);
    }

    // Quadratic term: chunk c's A-fragment is x_c (*) A_lin -> one v_pk_mul_bf16
    // per packed register. Fully unrolled so xr[c] stays in registers.
#pragma unroll
    for (int c = 0; c < N_FEAT; ++c) {
      unsigned uc = __builtin_bit_cast(unsigned, xr[c]);
      unsigned sp = __builtin_amdgcn_perm(uc, uc, 0x03020302u);  // splat bf16(x_c) to both halves
      v8u aqu;
#pragma unroll
      for (int r = 0; r < 8; ++r) {
        unsigned o;
        asm("v_pk_mul_bf16 %0, %1, %2" : "=v"(o) : "v"(au[r]), "v"(sp));
        aqu[r] = o;
      }
      v16bf aq  = __builtin_bit_cast(v16bf, aqu);
      v16bf bq0 = wbv[(2 + 2 * c) * 32 + lane];
      v16bf bq1 = wbv[(3 + 2 * c) * 32 + lane];
      acc0 = __builtin_amdgcn_wmma_f32_16x16x32_bf16(false, aq, false, bq0,
                                                     (short)0, acc0, false, false);
      acc1 = __builtin_amdgcn_wmma_f32_16x16x32_bf16(false, aq, false, bq1,
                                                     (short)0, acc1, false, false);
    }

    // Transpose D (C/D layout: VGPR r, lane -> M = r + 8*(lane>>4), N = lane&15)
    // into row-major LDS, then residual add back into the row registers.
#pragma unroll
    for (int r = 0; r < 8; ++r) {
      tw[(r + 8 * hlf) * N_FEAT + lrow]      = acc0[r];
      tw[(r + 8 * hlf) * N_FEAT + 16 + lrow] = acc1[r];
    }
    __syncthreads();
    const float4* ty = (const float4*)(tw + lrow * N_FEAT);
#pragma unroll
    for (int i = 0; i < 8; ++i) {
      float4 y = ty[i];
      xr[4 * i + 0] += y.x; xr[4 * i + 1] += y.y;
      xr[4 * i + 2] += y.z; xr[4 * i + 3] += y.w;
    }
    __syncthreads();
  }

  if (lane < 16) out[rowBase + lrow] = xr[0];  // N_TARGETS = 1
}

extern "C" void kernel_launch(void* const* d_in, const int* in_sizes, int n_in,
                              void* d_out, int out_size, void* d_ws, size_t ws_size,
                              hipStream_t stream) {
  const float* x = (const float*)d_in[0];   // (131072, 32) f32
  const float* W = (const float*)d_in[1];   // (1057, 32)  f32
  float* out = (float*)d_out;               // (131072, 1) f32
  unsigned short* wb = (unsigned short*)d_ws;  // 67584 B of bf16 B-fragments

  const int n = in_sizes[0] / N_FEAT;

  // One-time (per launch) repack of W into WMMA B-fragment order, bf16.
  tmpnn_pack_w<<<(PACK_ELEMS + 511) / 512, 512, 0, stream>>>(W, wb);

  // 8 waves/block * 16 rows/wave = 128 rows/block.
  const long ntiles = n >> 4;
  const int blocks = (int)((ntiles + 7) / 8);
  tmpnn_main<<<blocks, 256, 0, stream>>>(x, W, wb, out, n);
}
